// PoseGenerator_23733989278184
// MI455X (gfx1250) — compile-verified, matrix operand order checked by reading, not run on hardware
//
#include <hip/hip_runtime.h>

typedef __attribute__((ext_vector_type(16))) _Float16 v16h;
typedef __attribute__((ext_vector_type(8)))  float    v8f;

// ---------------------------------------------------------------------------
// Model dims
// ---------------------------------------------------------------------------
#define Bn   64
#define Tn   128
#define Dn   64
#define H1n  32
#define H2n  128
#define H3n  512
#define H4n  64
#define FBn  1024      // 2*H3
#define G3n  192       // 3*H4
#define NTOK 8192      // B*T

// ---------------------------------------------------------------------------
// f16 weight workspace offsets (elements)
// ---------------------------------------------------------------------------
constexpr size_t O_ENC_L2  = 0;                          // 4 x (128*64), order ra,la,rl,ll
constexpr size_t O_ENC_L3A = O_ENC_L2  + 4 * 8192;
constexpr size_t O_ENC_L3L = O_ENC_L3A + 131072;
constexpr size_t O_GRU_WIH = O_ENC_L3L + 131072;         // 192*1024
constexpr size_t O_GRU_WHH = O_GRU_WIH + 196608;         // 192*64
constexpr size_t O_CELL_WIH= O_GRU_WHH + 12288;
constexpr size_t O_CELL_WHH= O_CELL_WIH+ 12288;          // contiguous after wih
constexpr size_t O_DEC_L4  = O_CELL_WHH+ 12288;          // 1024*64
constexpr size_t O_DEC_L3A = O_DEC_L4  + 65536;          // 256*512
constexpr size_t O_DEC_L3L = O_DEC_L3A + 131072;
constexpr size_t O_DEC_L2  = O_DEC_L3L + 131072;         // 4 x (64*128), order ra,la,rl,ll
constexpr size_t F16_TOTAL = O_DEC_L2  + 4 * 8192;       // 888832 halves

// f32 workspace offsets (elements), base after f16 region
constexpr size_t O_FB  = 0;                              // 8192*1024
constexpr size_t O_GI  = O_FB  + (size_t)NTOK * FBn;     // 8192*192
constexpr size_t O_BNS = O_GI  + (size_t)NTOK * G3n;     // 1024 sums
constexpr size_t O_BNQ = O_BNS + 1024;                   // 1024 sumsq
constexpr size_t O_BSC = O_BNQ + 1024;                   // scale
constexpr size_t O_BSH = O_BSC + 1024;                   // shift
constexpr size_t O_ZP  = O_BSH + 1024;                   // 64*64
constexpr size_t O_ZG  = O_ZP  + 4096;
constexpr size_t O_QV  = O_ZG  + 4096;                   // 64*128*64
constexpr size_t O_LOSS= O_QV  + 524288;                 // 8 accumulators

// ---------------------------------------------------------------------------
// WMMA helpers (v_wmma_f32_16x16x32_f16)
// ---------------------------------------------------------------------------
__device__ inline v8f wmma32(v16h a, v16h b, v8f c) {
    return __builtin_amdgcn_wmma_f32_16x16x32_f16(false, a, false, b, (short)0, c,
                                                  false, false);
}

// Load a 16x32 f16 A/B tile. base points at (row0, k0). Lanes 0..15: row=lane,
// K chunks {0..7,16..23}; lanes 16..31: row=lane-16, K chunks {8..15,24..31}.
__device__ inline v16h load_tile_h(const _Float16* base, int ld) {
    int lane = threadIdx.x & 31;
    const _Float16* p = base + (lane & 15) * ld + ((lane >> 4) << 3);
    v16h v;
#pragma unroll
    for (int i = 0; i < 8; ++i) v[i] = p[i];
#pragma unroll
    for (int i = 0; i < 8; ++i) v[8 + i] = p[16 + i];
    return v;
}

// Same tile load from an f32 buffer, converting to f16 on the fly.
__device__ inline v16h load_tile_f(const float* base, int ld) {
    int lane = threadIdx.x & 31;
    const float* p = base + (lane & 15) * ld + ((lane >> 4) << 3);
    v16h v;
#pragma unroll
    for (int i = 0; i < 8; ++i) v[i] = (_Float16)p[i];
#pragma unroll
    for (int i = 0; i < 8; ++i) v[8 + i] = (_Float16)p[16 + i];
    return v;
}

__device__ inline float sigm(float v) { return 1.0f / (1.0f + __expf(-v)); }

// ---------------------------------------------------------------------------
// Utility kernels
// ---------------------------------------------------------------------------
__global__ void k_zero(float* p, int n) {
    int i = blockIdx.x * blockDim.x + threadIdx.x;
    if (i < n) p[i] = 0.f;
}

__global__ void k_cvt(const float* __restrict__ s, _Float16* __restrict__ d, int n) {
    int i = blockIdx.x * blockDim.x + threadIdx.x;
    if (i < n) d[i] = (_Float16)s[i];
}

// ---------------------------------------------------------------------------
// Encoder feature kernel: per 16-token tile, l1 (VALU) -> l2 -> l3 (WMMA),
// writes fb [NTOK][1024] and BN partial sums.
// ---------------------------------------------------------------------------
struct EncP {
    const float* x;                                    // [NTOK][64]
    const float *w1ra,*b1ra,*w1la,*b1la,*w1rl,*b1rl,*w1ll,*b1ll,*w1t,*b1t;
    const _Float16* w2[4]; const float* b2[4];          // order ra,la,rl,ll
    const _Float16 *w3a,*w3l; const float *b3a,*b3l;
    float *fb, *bnsum, *bnsq;
};

__global__ __launch_bounds__(256) void k_enc_feat(EncP p) {
    __shared__ float    Xt[16][64];
    __shared__ _Float16 A2[4][16][64];     // l2 inputs: [limb1 | mid1]
    __shared__ _Float16 A3[2][16][256];    // l3 inputs: up_in, lo_in
    int tid = threadIdx.x, wave = tid >> 5, lane = tid & 31;
    int tok0 = blockIdx.x * 16;

    for (int i = tid; i < 16 * 64; i += 256)
        Xt[i >> 6][i & 63] = p.x[(size_t)(tok0 + (i >> 6)) * 64 + (i & 63)];
    __syncthreads();

    // l1: 16 tokens x 5 parts x 32 outs
    for (int u = tid; u < 2560; u += 256) {
        int tok = u / 160, r = u - tok * 160, part = r >> 5, o = r & 31;
        float acc = 0.f, val;
        if (part == 0)      { const float* w = p.w1ra + o * 9;  for (int k = 0; k < 9;  ++k) acc += w[k] * Xt[tok][22 + k]; val = acc + p.b1ra[o]; }
        else if (part == 1) { const float* w = p.w1la + o * 9;  for (int k = 0; k < 9;  ++k) acc += w[k] * Xt[tok][13 + k]; val = acc + p.b1la[o]; }
        else if (part == 2) { const float* w = p.w1rl + o * 15; for (int k = 0; k < 15; ++k) acc += w[k] * Xt[tok][46 + k]; val = acc + p.b1rl[o]; }
        else if (part == 3) { const float* w = p.w1ll + o * 15; for (int k = 0; k < 15; ++k) acc += w[k] * Xt[tok][31 + k]; val = acc + p.b1ll[o]; }
        else { const float* w = p.w1t + o * 16;
               for (int k = 0; k < 13; ++k) acc += w[k] * Xt[tok][k];
               for (int k = 0; k < 3;  ++k) acc += w[13 + k] * Xt[tok][61 + k];
               val = acc + p.b1t[o]; }
        if (part < 4) A2[part][tok][o] = (_Float16)val;
        else {
            _Float16 hv = (_Float16)val;
            A2[0][tok][32 + o] = hv; A2[1][tok][32 + o] = hv;
            A2[2][tok][32 + o] = hv; A2[3][tok][32 + o] = hv;
        }
    }
    __syncthreads();

    // l2: 4 layers x (M=16, N=128, K=64) -> l3 inputs (f16)
    for (int t = wave; t < 32; t += 8) {
        int layer = t >> 3, ni = (t & 7) << 4;
        v8f acc = {};
        for (int k = 0; k < 64; k += 32) {
            v16h a  = load_tile_h(&A2[layer][0][0] + k, 64);
            v16h bm = load_tile_h(p.w2[layer] + ni * 64 + k, 64);
            acc = wmma32(a, bm, acc);
        }
        int col = ni + (lane & 15);
        float bias = p.b2[layer][col];
        int buf = layer >> 1, coff = (layer & 1) * 128;
#pragma unroll
        for (int v = 0; v < 8; ++v) {
            int row = v + ((lane >> 4) << 3);
            A3[buf][row][coff + col] = (_Float16)(acc[v] + bias);
        }
    }
    __syncthreads();

    // l3: 2 layers x (M=16, N=512, K=256) -> fb + BN partials
    for (int t = wave; t < 64; t += 8) {
        int layer = t >> 5, ni = (t & 31) << 4;
        const _Float16* W = layer ? p.w3l : p.w3a;
        const float* bias = layer ? p.b3l : p.b3a;
        v8f acc = {};
        for (int k = 0; k < 256; k += 32) {
            v16h a  = load_tile_h(&A3[layer][0][0] + k, 256);
            v16h bm = load_tile_h(W + ni * 256 + k, 256);
            acc = wmma32(a, bm, acc);
        }
        int col = ni + (lane & 15);
        float bv = bias[col];
        int ch = layer * 512 + col;
        float s = 0.f, sq = 0.f;
#pragma unroll
        for (int v = 0; v < 8; ++v) {
            int row = v + ((lane >> 4) << 3);
            float val = acc[v] + bv;
            p.fb[(size_t)(tok0 + row) * FBn + ch] = val;
            s += val; sq += val * val;
        }
        atomicAdd(&p.bnsum[ch], s);
        atomicAdd(&p.bnsq[ch], sq);
    }
}

// ---------------------------------------------------------------------------
// BN finalize: per-channel scale/shift
// ---------------------------------------------------------------------------
__global__ void k_bn(const float* bnsum, const float* bnsq, const float* gamma,
                     const float* beta, float* scale, float* shift) {
    int c = blockIdx.x * blockDim.x + threadIdx.x;
    if (c >= FBn) return;
    const float invN = 1.0f / (float)NTOK;
    float mu  = bnsum[c] * invN;
    float var = bnsq[c] * invN - mu * mu;
    float sc  = gamma[c] * rsqrtf(var + 1e-5f);
    scale[c] = sc;
    shift[c] = beta[c] - mu * sc;
}

// ---------------------------------------------------------------------------
// GI = BN(fb) @ Wih^T + bih  (no recurrence -> fully parallel, K=1024)
// ---------------------------------------------------------------------------
struct GiP {
    const float* fb; const float* scale; const float* shift;
    const _Float16* wih; const float* bih; float* gi;
};

__global__ __launch_bounds__(256) void k_gi(GiP p) {
    __shared__ _Float16 A[16][1024];
    int tid = threadIdx.x, wave = tid >> 5, lane = tid & 31;
    int tok0 = blockIdx.x * 16;
    for (int i = tid; i < 16 * 1024; i += 256) {
        int r = i >> 10, c = i & 1023;
        A[r][c] = (_Float16)(p.fb[(size_t)(tok0 + r) * FBn + c] * p.scale[c] + p.shift[c]);
    }
    __syncthreads();
    for (int t = wave; t < 12; t += 8) {           // N=192 -> 12 tiles
        int ni = t << 4;
        v8f acc = {};
        for (int k = 0; k < 1024; k += 32) {
            v16h a  = load_tile_h(&A[0][0] + k, 1024);
            v16h bm = load_tile_h(p.wih + (size_t)ni * 1024 + k, 1024);
            acc = wmma32(a, bm, acc);
        }
        int col = ni + (lane & 15);
        float bv = p.bih[col];
#pragma unroll
        for (int v = 0; v < 8; ++v) {
            int row = v + ((lane >> 4) << 3);
            p.gi[(size_t)(tok0 + row) * G3n + col] = acc[v] + bv;
        }
    }
}

// ---------------------------------------------------------------------------
// Encoder GRU scan: persistent single workgroup, h in LDS, T=128 steps.
// ---------------------------------------------------------------------------
struct GruP { const float* gi; const _Float16* whh; const float* bhh; float* zout; };

__global__ __launch_bounds__(256) void k_enc_gru(GruP p) {
    __shared__ float h[64][64];
    __shared__ float gh[64][192];
    int tid = threadIdx.x, wave = tid >> 5, lane = tid & 31;
    for (int i = tid; i < 4096; i += 256) h[i >> 6][i & 63] = 0.f;
    __syncthreads();
    for (int t = 0; t < Tn; ++t) {
        // keep the 24KB recurrent weight block warm in WGP$ (global_prefetch_b8)
        {
            const char* wp = (const char*)p.whh;
            for (size_t off = (size_t)tid * 128; off < 24576; off += 256 * 128)
                __builtin_prefetch(wp + off, 0, 0);
        }
        // gh = h @ whh^T + bhh : M=64(4) x N=192(12) tiles, K=64
        for (int tt = wave; tt < 48; tt += 8) {
            int mi = (tt / 12) << 4, ni = (tt % 12) << 4;
            v8f acc = {};
            for (int k = 0; k < 64; k += 32) {
                v16h a  = load_tile_f(&h[mi][k], 64);
                v16h bm = load_tile_h(p.whh + ni * 64 + k, 64);
                acc = wmma32(a, bm, acc);
            }
            int col = ni + (lane & 15);
            float bv = p.bhh[col];
#pragma unroll
            for (int v = 0; v < 8; ++v) {
                int row = mi + v + ((lane >> 4) << 3);
                gh[row][col] = acc[v] + bv;
            }
        }
        __syncthreads();
        for (int i = tid; i < 4096; i += 256) {
            int b = i >> 6, j = i & 63;
            const float* gir = &p.gi[(size_t)((b << 7) + t) * G3n];
            float r = sigm(gir[j] + gh[b][j]);
            float z = sigm(gir[64 + j] + gh[b][64 + j]);
            float n = tanhf(gir[128 + j] + r * gh[b][128 + j]);
            h[b][j] = (1.f - z) * n + z * h[b][j];
        }
        __syncthreads();
    }
    for (int i = tid; i < 4096; i += 256) p.zout[i] = h[i >> 6][i & 63];
}

// ---------------------------------------------------------------------------
// Velocity decoder scan: 2 blocks (z_p instance, language_z instance).
// Per step: GRU cell + dec_cell cascade, all GEMMs via WMMA, state in LDS.
// ---------------------------------------------------------------------------
struct DecP {
    const float *h0_a, *h0_b, *x0;                 // z_p, language_z, P_in
    const _Float16 *wih, *whh; const float *bih, *bhh;
    const _Float16* w4; const float* b4;
    const _Float16 *w3a, *w3l; const float *b3a, *b3l;
    const _Float16* w2[4]; const float* b2[4];      // ra, la, rl, ll
    const float *w1ra,*b1ra,*w1la,*b1la,*w1rl,*b1rl,*w1ll,*b1ll,*w1t,*b1t;
    float *out_a, *out_b;                           // Q_v ws, Q_v_lang (d_out)
};

// l1 LDS cache layout (floats)
#define W1_RA 0
#define W1_LA 288
#define W1_RL 576
#define W1_LL 1056
#define W1_T  1536
#define B1_RA 2048
#define B1_LA 2057
#define B1_RL 2066
#define B1_LL 2081
#define B1_T  2096
#define W1_TOT 2112

__global__ __launch_bounds__(256) void k_vel_dec(DecP p) {
    __shared__ float    h[64][64];
    __shared__ float    x[64][64];
    __shared__ float    scratch[64 * 384];          // gi|gh (f32), aliased by fb (f16)
    __shared__ _Float16 up[64][256];
    __shared__ _Float16 lo[64][256];
    __shared__ _Float16 l2o[4][64][64];             // ra2, la2, rl2, ll2
    __shared__ float    w1s[W1_TOT];                // l1 weights+biases (hot, serial path)

    int tid = threadIdx.x, wave = tid >> 5, lane = tid & 31;
    const float* h0 = blockIdx.x ? p.h0_b : p.h0_a;
    float* out      = blockIdx.x ? p.out_b : p.out_a;
    float* gig = scratch;
    float* ghg = scratch + 64 * 192;
    _Float16* fbh = (_Float16*)scratch;             // [64][512], lifetime-disjoint

    for (int i = tid; i < 4096; i += 256) {
        h[i >> 6][i & 63] = h0[i];
        x[i >> 6][i & 63] = p.x0[(size_t)(i >> 6) * (Tn * Dn) + (i & 63)];
    }
    // cache l1 heads in LDS: they sit on the 128-step serial critical path
    for (int i = tid; i < 288; i += 256) w1s[W1_RA + i] = p.w1ra[i];
    for (int i = tid; i < 288; i += 256) w1s[W1_LA + i] = p.w1la[i];
    for (int i = tid; i < 480; i += 256) w1s[W1_RL + i] = p.w1rl[i];
    for (int i = tid; i < 480; i += 256) w1s[W1_LL + i] = p.w1ll[i];
    for (int i = tid; i < 512; i += 256) w1s[W1_T  + i] = p.w1t[i];
    if (tid < 9)  w1s[B1_RA + tid] = p.b1ra[tid];
    if (tid < 9)  w1s[B1_LA + tid] = p.b1la[tid];
    if (tid < 15) w1s[B1_RL + tid] = p.b1rl[tid];
    if (tid < 15) w1s[B1_LL + tid] = p.b1ll[tid];
    if (tid < 16) w1s[B1_T  + tid] = p.b1t[tid];
    __syncthreads();

    for (int t = 0; t < Tn; ++t) {
        // keep the contiguous 48KB cell_wih|cell_whh block warm in WGP$
        {
            const char* wp = (const char*)p.wih;    // whh immediately follows in ws
            for (size_t off = (size_t)tid * 128; off < 49152; off += 256 * 128)
                __builtin_prefetch(wp + off, 0, 0);
        }
        // --- GRU gates: gi from x, gh from h (96 tiles, K=64) ---
        for (int tt = wave; tt < 96; tt += 8) {
            int which = (tt >= 48);
            int q = tt - which * 48;
            int mi = (q / 12) << 4, ni = (q % 12) << 4;
            const float* A       = which ? &h[0][0] : &x[0][0];
            const _Float16* W    = which ? p.whh : p.wih;
            const float* bias    = which ? p.bhh : p.bih;
            float* Og            = which ? ghg : gig;
            v8f acc = {};
            for (int k = 0; k < 64; k += 32) {
                v16h a  = load_tile_f(A + mi * 64 + k, 64);
                v16h bm = load_tile_h(W + ni * 64 + k, 64);
                acc = wmma32(a, bm, acc);
            }
            int col = ni + (lane & 15);
            float bv = bias[col];
#pragma unroll
            for (int v = 0; v < 8; ++v) {
                int row = mi + v + ((lane >> 4) << 3);
                Og[row * 192 + col] = acc[v] + bv;
            }
        }
        __syncthreads();
        // --- GRU elementwise ---
        for (int i = tid; i < 4096; i += 256) {
            int b = i >> 6, j = i & 63;
            float r = sigm(gig[b * 192 + j]      + ghg[b * 192 + j]);
            float z = sigm(gig[b * 192 + 64 + j] + ghg[b * 192 + 64 + j]);
            float n = tanhf(gig[b * 192 + 128 + j] + r * ghg[b * 192 + 128 + j]);
            h[b][j] = (1.f - z) * n + z * h[b][j];
        }
        __syncthreads();
        // --- fb = h @ w4^T + b4 (1024 out, in 2 halves) then l3 ---
        for (int half = 0; half < 2; ++half) {
            for (int tt = wave; tt < 128; tt += 8) {       // M=64(4) x N=512(32)
                int mi = (tt >> 5) << 4, ni = (tt & 31) << 4;
                v8f acc = {};
                for (int k = 0; k < 64; k += 32) {
                    v16h a  = load_tile_f(&h[mi][k], 64);
                    v16h bm = load_tile_h(p.w4 + (size_t)(half * 512 + ni) * 64 + k, 64);
                    acc = wmma32(a, bm, acc);
                }
                int col = ni + (lane & 15);
                float bv = p.b4[half * 512 + col];
#pragma unroll
                for (int v = 0; v < 8; ++v) {
                    int row = mi + v + ((lane >> 4) << 3);
                    fbh[row * 512 + col] = (_Float16)(acc[v] + bv);
                }
            }
            __syncthreads();
            _Float16* dst     = half ? &lo[0][0] : &up[0][0];
            const _Float16* W = half ? p.w3l : p.w3a;
            const float* bias = half ? p.b3l : p.b3a;
            for (int tt = wave; tt < 64; tt += 8) {        // M=64(4) x N=256(16), K=512
                int mi = (tt >> 4) << 4, ni = (tt & 15) << 4;
                v8f acc = {};
                for (int k = 0; k < 512; k += 32) {
                    v16h a  = load_tile_h(fbh + mi * 512 + k, 512);
                    v16h bm = load_tile_h(W + (size_t)ni * 512 + k, 512);
                    acc = wmma32(a, bm, acc);
                }
                int col = ni + (lane & 15);
                float bv = bias[col];
#pragma unroll
                for (int v = 0; v < 8; ++v) {
                    int row = mi + v + ((lane >> 4) << 3);
                    dst[row * 256 + col] = (_Float16)(acc[v] + bv);
                }
            }
            __syncthreads();
        }
        // --- l2: 4 layers, M=64(4) x N=64(4), K=128 ---
        for (int tt = wave; tt < 64; tt += 8) {
            int layer = tt >> 4, q = tt & 15, mi = (q >> 2) << 4, ni = (q & 3) << 4;
            const _Float16* A = (layer < 2) ? &up[0][0] : &lo[0][0];
            int coff = (layer & 1) * 128;
            v8f acc = {};
            for (int k = 0; k < 128; k += 32) {
                v16h a  = load_tile_h(A + mi * 256 + coff + k, 256);
                v16h bm = load_tile_h(p.w2[layer] + ni * 128 + k, 128);
                acc = wmma32(a, bm, acc);
            }
            int col = ni + (lane & 15);
            float bv = p.b2[layer][col];
#pragma unroll
            for (int v = 0; v < 8; ++v) {
                int row = mi + v + ((lane >> 4) << 3);
                l2o[layer][row][col] = (_Float16)(acc[v] + bv);
            }
        }
        __syncthreads();
        // --- l1 heads (LDS weights) + residual update + output write ---
        for (int u = tid; u < 4096; u += 256) {
            int b = u >> 6, unit = u & 63;
            float acc = 0.f, val; int pos;
            if (unit < 48) {
                int o, li, wo, bo;
                if (unit < 9)       { li = 0; o = unit;      wo = W1_RA; bo = B1_RA; pos = 22 + o; }
                else if (unit < 18) { li = 1; o = unit - 9;  wo = W1_LA; bo = B1_LA; pos = 13 + o; }
                else if (unit < 33) { li = 2; o = unit - 18; wo = W1_RL; bo = B1_RL; pos = 46 + o; }
                else                { li = 3; o = unit - 33; wo = W1_LL; bo = B1_LL; pos = 31 + o; }
                for (int k = 0; k < 32; ++k) acc += w1s[wo + o * 32 + k] * (float)l2o[li][b][k];
                val = acc + w1s[bo + o];
            } else {
                int j = unit - 48;
                for (int li = 0; li < 4; ++li)
                    for (int k = 0; k < 32; ++k)
                        acc += w1s[W1_T + j * 32 + k] * (float)l2o[li][b][32 + k];
                val = acc * 0.25f + w1s[B1_T + j];
                pos = (j < 13) ? j : (48 + j);              // 61 + (j-13)
            }
            float nx = x[b][pos] + val;
            x[b][pos] = nx;
            out[(size_t)(b * Tn + t) * Dn + pos] = nx;
        }
        __syncthreads();
    }
}

// ---------------------------------------------------------------------------
// Loss kernels
// ---------------------------------------------------------------------------
__global__ __launch_bounds__(256) void k_huber(const float* __restrict__ a,
                                               const float* __restrict__ b,
                                               int n, float* accum) {
    float s = 0.f;
    for (int i = blockIdx.x * blockDim.x + threadIdx.x; i < n;
         i += gridDim.x * blockDim.x) {
        float d = a[i] - b[i];
        float ad = fabsf(d);
        s += (ad < 1.f) ? 0.5f * d * d : ad - 0.5f;
    }
    __shared__ float red[256];
    red[threadIdx.x] = s;
    __syncthreads();
    for (int off = 128; off; off >>= 1) {
        if (threadIdx.x < off) red[threadIdx.x] += red[threadIdx.x + off];
        __syncthreads();
    }
    if (threadIdx.x == 0) atomicAdd(accum, red[0]);
}

__global__ __launch_bounds__(256) void k_huber_vel(const float* __restrict__ P,
                                                   const float* __restrict__ Q,
                                                   float* accum) {
    const int n = Bn * (Tn - 1) * Dn;
    float s = 0.f;
    for (int i = blockIdx.x * blockDim.x + threadIdx.x; i < n;
         i += gridDim.x * blockDim.x) {
        int b = i / ((Tn - 1) * Dn);
        int r = i - b * (Tn - 1) * Dn;
        int t = r / Dn, dd = r - t * Dn;
        size_t idx = (size_t)(b * Tn + t) * Dn + dd;
        float va = P[idx + Dn] - P[idx];
        float vb = Q[idx + Dn] - Q[idx];
        float d = va - vb;
        float ad = fabsf(d);
        s += (ad < 1.f) ? 0.5f * d * d : ad - 0.5f;
    }
    __shared__ float red[256];
    red[threadIdx.x] = s;
    __syncthreads();
    for (int off = 128; off; off >>= 1) {
        if (threadIdx.x < off) red[threadIdx.x] += red[threadIdx.x + off];
        __syncthreads();
    }
    if (threadIdx.x == 0) atomicAdd(accum, red[0]);
}

__global__ void k_finalize(const float* acc, float* out) {
    if (blockIdx.x == 0 && threadIdx.x == 0) {
        out[0] = 0.001f * (acc[0] / 4096.f);                         // manifold
        out[1] = 0.1f   * (acc[1] / 4096.f);                         // encoder
        out[2] = acc[2] / 524288.f;                                  // recon
        out[3] = acc[3] / 524288.f;                                  // recon_lang
        out[4] = 0.1f * (acc[4] / 520192.f + acc[5] / 520192.f);     // velocity
    }
}

// ---------------------------------------------------------------------------
// Host driver
// ---------------------------------------------------------------------------
extern "C" void kernel_launch(void* const* d_in, const int* in_sizes, int n_in,
                              void* d_out, int out_size, void* d_ws, size_t ws_size,
                              hipStream_t stream) {
    (void)in_sizes; (void)n_in; (void)out_size; (void)ws_size;
    auto F = [&](int i) { return (const float*)d_in[i]; };
    // flatten order assumed: P_in, language_z, then params via sorted-key tree
    // flatten ('dec' < 'enc'; within each linear 'b' < 'w').
    _Float16* W16 = (_Float16*)d_ws;
    float* FS = (float*)((char*)d_ws + F16_TOTAL * sizeof(_Float16));

    auto cvt = [&](int src, size_t dstOff, int n) {
        k_cvt<<<(n + 255) / 256, 256, 0, stream>>>(F(src), W16 + dstOff, n);
    };
    // enc l2 (ra=51, la=47, rl=53, ll=49), l3 (arm=55, leg=57)
    cvt(51, O_ENC_L2 + 0 * 8192, 8192);
    cvt(47, O_ENC_L2 + 1 * 8192, 8192);
    cvt(53, O_ENC_L2 + 2 * 8192, 8192);
    cvt(49, O_ENC_L2 + 3 * 8192, 8192);
    cvt(55, O_ENC_L3A, 131072);
    cvt(57, O_ENC_L3L, 131072);
    cvt(35, O_GRU_WIH, 196608);
    cvt(34, O_GRU_WHH, 12288);
    cvt(5,  O_CELL_WIH, 12288);
    cvt(4,  O_CELL_WHH, 12288);
    cvt(29, O_DEC_L4, 65536);
    cvt(25, O_DEC_L3A, 131072);
    cvt(27, O_DEC_L3L, 131072);
    cvt(21, O_DEC_L2 + 0 * 8192, 8192);  // dec l2_rarm
    cvt(17, O_DEC_L2 + 1 * 8192, 8192);  // larm
    cvt(23, O_DEC_L2 + 2 * 8192, 8192);  // rleg
    cvt(19, O_DEC_L2 + 3 * 8192, 8192);  // lleg

    k_zero<<<8, 256, 0, stream>>>(FS + O_BNS, 2048);
    k_zero<<<1, 256, 0, stream>>>(FS + O_LOSS, 8);

    EncP ep{};
    ep.w1ra = F(41); ep.b1ra = F(40);
    ep.w1la = F(37); ep.b1la = F(36);
    ep.w1rl = F(43); ep.b1rl = F(42);
    ep.w1ll = F(39); ep.b1ll = F(38);
    ep.w1t  = F(45); ep.b1t  = F(44);
    ep.w2[0] = W16 + O_ENC_L2;             ep.b2[0] = F(50);
    ep.w2[1] = W16 + O_ENC_L2 + 8192;      ep.b2[1] = F(46);
    ep.w2[2] = W16 + O_ENC_L2 + 16384;     ep.b2[2] = F(52);
    ep.w2[3] = W16 + O_ENC_L2 + 24576;     ep.b2[3] = F(48);
    ep.w3a = W16 + O_ENC_L3A; ep.b3a = F(54);
    ep.w3l = W16 + O_ENC_L3L; ep.b3l = F(56);
    ep.fb = FS + O_FB; ep.bnsum = FS + O_BNS; ep.bnsq = FS + O_BNQ;

    GiP gp{FS + O_FB, FS + O_BSC, FS + O_BSH, W16 + O_GRU_WIH, F(33), FS + O_GI};

    // ---- encoder pass A: pose_enc(P_in) -> z_p
    ep.x = F(0);
    k_enc_feat<<<512, 256, 0, stream>>>(ep);
    k_bn<<<4, 256, 0, stream>>>(FS + O_BNS, FS + O_BNQ, F(31), F(30),
                                FS + O_BSC, FS + O_BSH);
    k_gi<<<512, 256, 0, stream>>>(gp);
    GruP rp{FS + O_GI, W16 + O_GRU_WHH, F(32), FS + O_ZP};
    k_enc_gru<<<1, 256, 0, stream>>>(rp);

    // ---- velocity decoders (z_p instance + language_z instance)
    DecP dp{};
    dp.h0_a = FS + O_ZP; dp.h0_b = F(1); dp.x0 = F(0);
    dp.wih = W16 + O_CELL_WIH; dp.whh = W16 + O_CELL_WHH;
    dp.bih = F(3); dp.bhh = F(2);
    dp.w4 = W16 + O_DEC_L4; dp.b4 = F(28);
    dp.w3a = W16 + O_DEC_L3A; dp.b3a = F(24);
    dp.w3l = W16 + O_DEC_L3L; dp.b3l = F(26);
    dp.w2[0] = W16 + O_DEC_L2;         dp.b2[0] = F(20);
    dp.w2[1] = W16 + O_DEC_L2 + 8192;  dp.b2[1] = F(16);
    dp.w2[2] = W16 + O_DEC_L2 + 16384; dp.b2[2] = F(22);
    dp.w2[3] = W16 + O_DEC_L2 + 24576; dp.b2[3] = F(18);
    dp.w1ra = F(11); dp.b1ra = F(10);
    dp.w1la = F(7);  dp.b1la = F(6);
    dp.w1rl = F(13); dp.b1rl = F(12);
    dp.w1ll = F(9);  dp.b1ll = F(8);
    dp.w1t  = F(15); dp.b1t  = F(14);
    dp.out_a = FS + O_QV; dp.out_b = (float*)d_out;
    k_vel_dec<<<2, 256, 0, stream>>>(dp);

    // ---- encoder pass B: pose_enc(Q_v) -> z_gen
    k_zero<<<8, 256, 0, stream>>>(FS + O_BNS, 2048);
    ep.x = FS + O_QV;
    k_enc_feat<<<512, 256, 0, stream>>>(ep);
    k_bn<<<4, 256, 0, stream>>>(FS + O_BNS, FS + O_BNQ, F(31), F(30),
                                FS + O_BSC, FS + O_BSH);
    k_gi<<<512, 256, 0, stream>>>(gp);
    GruP rp2{FS + O_GI, W16 + O_GRU_WHH, F(32), FS + O_ZG};
    k_enc_gru<<<1, 256, 0, stream>>>(rp2);

    // ---- losses
    float* acc = FS + O_LOSS;
    k_huber<<<16, 256, 0, stream>>>(FS + O_ZG, FS + O_ZP, 4096, acc + 0);
    k_huber<<<16, 256, 0, stream>>>(FS + O_ZP, F(1), 4096, acc + 1);
    k_huber<<<256, 256, 0, stream>>>(FS + O_QV, F(0), 524288, acc + 2);
    k_huber<<<256, 256, 0, stream>>>((const float*)d_out, F(0), 524288, acc + 3);
    k_huber_vel<<<256, 256, 0, stream>>>(F(0), FS + O_QV, acc + 4);
    k_huber_vel<<<256, 256, 0, stream>>>(F(0), (const float*)d_out, acc + 5);
    k_finalize<<<1, 64, 0, stream>>>(acc, (float*)d_out + 524288);
}
